// LatentAttention_27616639713601
// MI455X (gfx1250) — compile-verified, matrix operand order checked by reading, not run on hardware
//
#include <hip/hip_runtime.h>
#include <math.h>

// Problem constants (from reference)
#define SEQ    16384
#define HID    4096
#define NHEADS 32
#define RLAT   64
#define DH     128      // HID / NHEADS

typedef __bf16 bf16_t;
typedef __attribute__((ext_vector_type(4)))  bf16_t v4bf;
typedef __attribute__((ext_vector_type(8)))  bf16_t v8bf;
typedef __attribute__((ext_vector_type(16))) bf16_t v16bf;
typedef __attribute__((ext_vector_type(8)))  float  v8f;
typedef __attribute__((ext_vector_type(4)))  unsigned int v4u;
typedef __attribute__((ext_vector_type(8)))  int    v8i;
typedef __attribute__((ext_vector_type(4)))  int    v4i;

// GEMM tiling: 256 threads = 8 waves (wave32). Block tile 128x128, K-step 32.
// Waves 4(m) x 2(n): each wave owns 32x64 = 2x4 subtiles of 16x16.
#define BM  128
#define BN  128
#define BK  32
#define LDA 40   // 32 + 8 pad (bf16) == TDM pad: 16 DWORDs data + 4 DWORDs pad
#define LDB 40

__device__ __forceinline__ float gelu_exact(float x) {
    return 0.5f * x * (1.0f + erff(x * 0.70710678118654752f));
}

// ---------------------------------------------------------------------------
// TDM: async 2-D tile load (bf16) global -> LDS with built-in row padding.
// Descriptor packed per CDNA5 ISA 8.3/8.4:
//   g0: count=1 | lds_addr | global_addr[56:0] | type=2
//   g1: data_size=1(2B), pad_enable, pad_interval=3 (16 DWORDs),
//       pad_amount=3 (4 DWORDs)  -> LDS row stride = 40 bf16 = LDA/LDB
//       tensor_dim0=HID, tensor_dim0_stride=HID, tile_dim0=BK, tile_dim1=128
// ---------------------------------------------------------------------------
__device__ __forceinline__ void tdm_load_tile(const bf16_t* gptr, void* lptr,
                                              unsigned tensor_d1) {
    const unsigned long long ga = (unsigned long long)(uintptr_t)gptr;
    const unsigned la = (unsigned)(uintptr_t)lptr;   // low 32 bits = LDS offset
    v4u g0;
    g0[0] = 1u;                                   // count=1, user mode
    g0[1] = la;                                   // lds_addr
    g0[2] = (unsigned)ga;                         // global_addr[31:0]
    g0[3] = (unsigned)((ga >> 32) & 0x1FFFFFFu) | (2u << 30);  // addr[56:32]|type=2
    v8i g1;
    g1[0] = (int)((1u << 16) |                    // data_size = 1 (2 bytes)
                  (1u << 20) |                    // pad_enable
                  (3u << 22) |                    // pad_interval: 16 DWORDs
                  (3u << 25));                    // pad_amount : 4 DWORDs
    g1[1] = (int)(((unsigned)HID & 0xFFFFu) << 16);              // tensor_dim0 lo
    g1[2] = (int)(((unsigned)HID >> 16) | ((tensor_d1 & 0xFFFFu) << 16));
    g1[3] = (int)((tensor_d1 >> 16) | ((unsigned)BK << 16));     // tile_dim0
    g1[4] = (int)BM;                              // tile_dim1 = 128 rows, dim2=0
    g1[5] = (int)HID;                             // tensor_dim0_stride[31:0]
    g1[6] = 0;
    g1[7] = 0;
    v4i z = {0, 0, 0, 0};
#if defined(__clang_major__) && (__clang_major__ >= 23)
    v8i z8 = {0, 0, 0, 0, 0, 0, 0, 0};
    __builtin_amdgcn_tensor_load_to_lds(g0, g1, z, z, z8, 0);
#else
    __builtin_amdgcn_tensor_load_to_lds(g0, g1, z, z, 0);
#endif
}

// ---------------------------------------------------------------------------
// GEMM: C[M,N] = A[M,K](bf16, row-major) * B(bf16, PRE-TRANSPOSED [N][K])
// EPI: 0 = *scale   1 = plain   2 = bias + exact gelu   3 = bias + residual
// WF/WB: write f32 / bf16 outputs.
// ---------------------------------------------------------------------------
template<int EPI, bool WF, bool WB>
__global__ __launch_bounds__(256)
void gemm_tdm(const bf16_t* __restrict__ A, const bf16_t* __restrict__ B,
              float* __restrict__ Cf, bf16_t* __restrict__ Cb,
              const float* __restrict__ bias, const float* __restrict__ res,
              float scale) {
    __shared__ bf16_t As[2][BM * LDA];
    __shared__ bf16_t Bs[2][BN * LDB];

    const int tid  = threadIdx.x;
    const int lane = tid & 31;
    const int wv   = tid >> 5;
    const int wm   = wv & 3;          // wave-row: 32 rows each
    const int wn   = wv >> 2;         // wave-col: 64 cols each
    const int hf   = lane >> 4;       // K-group select per ISA 16-bit layout
    const int hl   = lane & 15;

    const int bm = blockIdx.y * BM;
    const int bn = blockIdx.x * BN;
    const bf16_t* Ag = A + (size_t)bm * HID;
    const bf16_t* Bg = B + (size_t)bn * HID;

    const int KT = HID / BK;          // 128 K-stages

    // prologue: prefetch stages 0 and 1 via TDM (wave 0 issues, per-wave op)
    if (wv == 0) {
        tdm_load_tile(Ag + 0,  &As[0][0], SEQ);
        tdm_load_tile(Bg + 0,  &Bs[0][0], HID);
        tdm_load_tile(Ag + BK, &As[1][0], SEQ);
        tdm_load_tile(Bg + BK, &Bs[1][0], HID);
    }

    v8f acc[2][4] = {};

    #pragma unroll 2
    for (int kt = 0; kt < KT; ++kt) {
        if (wv == 0) {
            if (kt + 1 < KT) __builtin_amdgcn_s_wait_tensorcnt(2);
            else             __builtin_amdgcn_s_wait_tensorcnt(0);
        }
        __syncthreads();                       // publish stage kt tiles

        const bf16_t* Ab = &As[kt & 1][0];
        const bf16_t* Bb = &Bs[kt & 1][0];

        // fragment loads per CDNA5 16-bit A layout:
        // lane<16: K={0..7,16..23}; lane>=16: K={8..15,24..31}
        v16bf aF[2], bF[4];
        #pragma unroll
        for (int mt = 0; mt < 2; ++mt) {
            const bf16_t* p = Ab + (wm * 32 + mt * 16 + hl) * LDA + hf * 8;
            v8bf lo = *(const v8bf*)(p);
            v8bf hi = *(const v8bf*)(p + 16);
            aF[mt] = __builtin_shufflevector(lo, hi,
                      0,1,2,3,4,5,6,7,8,9,10,11,12,13,14,15);
        }
        #pragma unroll
        for (int nt = 0; nt < 4; ++nt) {
            const bf16_t* p = Bb + (wn * 64 + nt * 16 + hl) * LDB + hf * 8;
            v8bf lo = *(const v8bf*)(p);
            v8bf hi = *(const v8bf*)(p + 16);
            bF[nt] = __builtin_shufflevector(lo, hi,
                      0,1,2,3,4,5,6,7,8,9,10,11,12,13,14,15);
        }
        #pragma unroll
        for (int mt = 0; mt < 2; ++mt)
            #pragma unroll
            for (int nt = 0; nt < 4; ++nt)
                acc[mt][nt] = __builtin_amdgcn_wmma_f32_16x16x32_bf16(
                    false, aF[mt], false, bF[nt],
                    (short)0, acc[mt][nt], false, false);

        __syncthreads();                       // all waves done with buf kt&1
        if (wv == 0 && kt + 2 < KT) {
            const int k2 = (kt + 2) * BK;
            tdm_load_tile(Ag + k2, &As[kt & 1][0], SEQ);
            tdm_load_tile(Bg + k2, &Bs[kt & 1][0], HID);
        }
    }

    // epilogue: C layout is VGPR g -> row g + 8*hf, col = hl
    #pragma unroll
    for (int mt = 0; mt < 2; ++mt) {
        #pragma unroll
        for (int nt = 0; nt < 4; ++nt) {
            const int col = bn + wn * 64 + nt * 16 + hl;
            #pragma unroll
            for (int g = 0; g < 8; ++g) {
                const int row = bm + wm * 32 + mt * 16 + hf * 8 + g;
                float v = acc[mt][nt][g];
                if (EPI == 0) v *= scale;
                if (EPI == 2) v = gelu_exact(v + bias[col]);
                if (EPI == 3) v += bias[col] + res[(size_t)row * HID + col];
                if (WF) Cf[(size_t)row * HID + col] = v;
                if (WB) Cb[(size_t)row * HID + col] = (bf16_t)v;
            }
        }
    }
}

// ---------------------------------------------------------------------------
// Weight transpose+convert: f32 [K][N] -> bf16 [N][K] (32x32 LDS tiles)
// ---------------------------------------------------------------------------
__global__ __launch_bounds__(256)
void transpose_convert(const float* __restrict__ src, bf16_t* __restrict__ dst) {
    __shared__ float t[32][33];
    const int c0 = blockIdx.x * 32, r0 = blockIdx.y * 32;
    const int cc = threadIdx.x & 31, rr = threadIdx.x >> 5;   // rr 0..7
    #pragma unroll
    for (int i = 0; i < 4; ++i)
        t[rr + i * 8][cc] = src[(size_t)(r0 + rr + i * 8) * HID + c0 + cc];
    __syncthreads();
    #pragma unroll
    for (int i = 0; i < 4; ++i)
        dst[(size_t)(c0 + rr + i * 8) * HID + r0 + cc] = (bf16_t)t[cc][rr + i * 8];
}

// flat f32 -> bf16 convert (for the query activation)
__global__ __launch_bounds__(256)
void convert_bf16(const float* __restrict__ src, bf16_t* __restrict__ dst) {
    const size_t i = ((size_t)blockIdx.x * 256 + threadIdx.x) * 4;
    const float4 v = *(const float4*)(src + i);
    v4bf p;
    p[0] = (bf16_t)v.x; p[1] = (bf16_t)v.y;
    p[2] = (bf16_t)v.z; p[3] = (bf16_t)v.w;
    *(v4bf*)(dst + i) = p;
}

// ---------------------------------------------------------------------------
// Latent attention: block = one head h, 64 seq rows (8 waves x 8 rows).
// kv head slice (64x128 f32) in LDS (+1 pad). Reads xq f32, writes bf16.
// ---------------------------------------------------------------------------
__global__ __launch_bounds__(256)
void latent_attn(const float* __restrict__ xq, const float* __restrict__ kvl,
                 bf16_t* __restrict__ outb) {
    __shared__ float kvs[RLAT][DH + 1];
    const int tid  = threadIdx.x;
    const int lane = tid & 31;
    const int wv   = tid >> 5;
    const int h    = blockIdx.y;
    const int s0   = blockIdx.x * 64;

    #pragma unroll
    for (int i = 0; i < 32; ++i) {
        const int idx = tid + i * 256;
        const int r = idx >> 7, k = idx & 127;
        kvs[r][k] = kvl[(size_t)r * HID + h * DH + k];
    }
    __syncthreads();

    const int r0 = lane, r1 = lane + 32;

    for (int rr = 0; rr < 8; ++rr) {
        const int s = s0 + wv * 8 + rr;
        float xv[4];
        #pragma unroll
        for (int i = 0; i < 4; ++i)
            xv[i] = xq[(size_t)s * HID + h * DH + lane + i * 32];

        float sc0 = 0.f, sc1 = 0.f;
        #pragma unroll 4
        for (int k = 0; k < DH; ++k) {
            const float q = __shfl(xv[k >> 5], k & 31, 32);
            sc0 += q * kvs[r0][k];
            sc1 += q * kvs[r1][k];
        }
        float m = fmaxf(sc0, sc1);
        #pragma unroll
        for (int off = 16; off > 0; off >>= 1)
            m = fmaxf(m, __shfl_xor(m, off, 32));
        const float e0 = __expf(sc0 - m), e1 = __expf(sc1 - m);
        float sum = e0 + e1;
        #pragma unroll
        for (int off = 16; off > 0; off >>= 1)
            sum += __shfl_xor(sum, off, 32);
        const float inv = 1.0f / sum;
        const float p0 = e0 * inv, p1 = e1 * inv;

        float acc[4] = {0.f, 0.f, 0.f, 0.f};
        #pragma unroll 4
        for (int r = 0; r < RLAT; ++r) {
            const float pr = (r < 32) ? __shfl(p0, r, 32) : __shfl(p1, r - 32, 32);
            #pragma unroll
            for (int i = 0; i < 4; ++i)
                acc[i] += pr * kvs[r][lane + i * 32];
        }
        #pragma unroll
        for (int i = 0; i < 4; ++i)
            outb[(size_t)s * HID + h * DH + lane + i * 32] = (bf16_t)acc[i];
    }
}

extern "C" void kernel_launch(void* const* d_in, const int* in_sizes, int n_in,
                              void* d_out, int out_size, void* d_ws, size_t ws_size,
                              hipStream_t stream) {
    const float* query = (const float*)d_in[0];
    const float* kvl   = (const float*)d_in[1];
    const float* wq    = (const float*)d_in[2];
    const float* wo    = (const float*)d_in[3];
    const float* w1    = (const float*)d_in[4];
    const float* b1    = (const float*)d_in[5];
    const float* w2    = (const float*)d_in[6];
    const float* b2    = (const float*)d_in[7];
    float* out = (float*)d_out;

    const size_t wElems = (size_t)HID * HID;   // 16M
    const size_t aElems = (size_t)SEQ * HID;   // 64M
    char* ws = (char*)d_ws;
    bf16_t* wT   = (bf16_t*)ws;                         ws += wElems * 2;  //  32MB (reused per GEMM)
    float*  f32b = (float*)ws;                          ws += aElems * 4;  // 256MB (xq -> out2)
    bf16_t* bfA  = (bf16_t*)ws;                         ws += aElems * 2;  // 128MB (query_bf -> attn_bf -> h_bf)
    bf16_t* bfB  = (bf16_t*)ws;                                            // 128MB (out2_bf)

    dim3 blk(256);
    dim3 grd(HID / BN, SEQ / BM);              // 32 x 128 blocks
    dim3 grdT(HID / 32, HID / 32);             // weight transpose
    const int grdC = (int)(aElems / 1024);     // flat convert, 4 elems/thread
    const float scale = 0.125f;                // R^-0.5, R = 64

    // stage 0: bf16 operand prep
    transpose_convert<<<grdT, blk, 0, stream>>>(wq, wT);
    convert_bf16<<<grdC, blk, 0, stream>>>(query, bfA);
    // xq = (query @ wq) * scale   -> f32
    gemm_tdm<0, true, false><<<grd, blk, 0, stream>>>(bfA, wT, f32b, nullptr,
                                                      nullptr, nullptr, scale);
    transpose_convert<<<grdT, blk, 0, stream>>>(wo, wT);
    // per-head latent attention -> bf16 (A of next GEMM)
    latent_attn<<<dim3(SEQ / 64, NHEADS), blk, 0, stream>>>(f32b, kvl, bfA);
    // out2 = attn @ wo  -> f32 (residual) + bf16 (A of GEMM3)
    gemm_tdm<1, true, true><<<grd, blk, 0, stream>>>(bfA, wT, f32b, bfB,
                                                     nullptr, nullptr, 1.0f);
    transpose_convert<<<grdT, blk, 0, stream>>>(w1, wT);
    // h = gelu(out2 @ w1 + b1) -> bf16 only
    gemm_tdm<2, false, true><<<grd, blk, 0, stream>>>(bfB, wT, nullptr, bfA,
                                                      b1, nullptr, 1.0f);
    transpose_convert<<<grdT, blk, 0, stream>>>(w2, wT);
    // result = h @ w2 + b2 + out2 -> f32 output
    gemm_tdm<3, true, false><<<grd, blk, 0, stream>>>(bfA, wT, out, nullptr,
                                                      b2, f32b, 1.0f);
}
